// CausalSelfAttention_60413009986022
// MI455X (gfx1250) — matrix operand order repaired and checked by reading, not compiled, not run on hardware
//
#include <hip/hip_runtime.h>
#include <hip/hip_bf16.h>

typedef __attribute__((ext_vector_type(16))) _Float16 v16h;
typedef __attribute__((ext_vector_type(8)))  float    v8f;

constexpr int HDIM  = 768;
constexpr int NHEAD = 12;
constexpr int HD    = 64;
constexpr int TSEQ  = 2048;
constexpr int BATCH = 4;
#define NEGBIG (-1e9f)

union Frag16 { v16h h; uint4 u[2]; };
union Pack4  { _Float16 h[4]; uint2 u; };
union Pack8  { _Float16 h[8]; uint4 u; };

__device__ __forceinline__ v8f wmma_f16(v16h a, v16h b, v8f c) {
  return __builtin_amdgcn_wmma_f32_16x16x32_f16(false, a, false, b,
                                                (short)0, c, false, false);
}

// CDNA5 async copy: global -> LDS, tracked by ASYNCcnt (§10.7 / §15.18.3)
__device__ __forceinline__ void async_load_b128(uint32_t lds_addr,
                                                const void* gptr) {
  asm volatile("global_load_async_to_lds_b128 %0, %1, off"
               :: "v"(lds_addr), "v"((unsigned long long)(uintptr_t)gptr)
               : "memory");
}
__device__ __forceinline__ void wait_async0() {
  asm volatile("s_wait_asynccnt 0" ::: "memory");
}
__device__ __forceinline__ uint32_t lds_addr_of(const void* p) {
  return (uint32_t)(uintptr_t)p;  // LDS aperture: addr[31:0] is the LDS offset
}

// ---------------------------------------------------------------------------
// Kernel 1: C = X * W^T + bias (f32 in, f16 out in attention layouts)
//   kind 0/1 -> outQK[B][NH][T][HD];  kind 2 -> outVt[B][NH][HD][T]
// 256 thr (8 waves), tile 128(M) x 64(N), BK = 64 (two WMMA K-steps/stage)
// ---------------------------------------------------------------------------
__global__ __launch_bounds__(256)
void qkv_proj_kernel(const float* __restrict__ X,
                     const float* __restrict__ W,
                     const float* __restrict__ bias,
                     _Float16* __restrict__ outQK,
                     _Float16* __restrict__ outVt,
                     int kind)
{
  __shared__ _Float16 sX[128 * 72];  // 128 x 64 (+8 pad)
  __shared__ _Float16 sW[ 64 * 72];

  const int tid  = threadIdx.x;
  const int lane = tid & 31;
  const int wave = tid >> 5;
  const int wm   = wave >> 1;      // 0..3 : 32-row band
  const int wn   = wave & 1;       // 0..1 : 32-col band
  const int lh   = lane >> 4;
  const int l16  = lane & 15;

  const int row0 = blockIdx.y * 128;   // flattened B*T
  const int col0 = blockIdx.x * 64;    // output feature

  // fixed per-thread staging coordinates (trip counts are compile-time)
  const int xm = tid >> 4, xk4 = tid & 15;       // X: 16 rows apart per step
  const int wmr = tid >> 4;                      // W: 16 rows apart per step

  const v8f vzero = {};
  v8f acc[2][2];
  acc[0][0] = vzero; acc[0][1] = vzero; acc[1][0] = vzero; acc[1][1] = vzero;

  for (int k0 = 0; k0 < HDIM; k0 += 64) {
    // ---- batch all global loads first (12 x b128), then convert + store ----
    float4 xf[8], wf[4];
#pragma unroll
    for (int ii = 0; ii < 8; ii++) {
      const int m = xm + ii * 16;
      xf[ii] = *(const float4*)&X[(size_t)(row0 + m) * HDIM + k0 + xk4 * 4];
    }
#pragma unroll
    for (int ii = 0; ii < 4; ii++) {
      const int n = wmr + ii * 16;
      wf[ii] = *(const float4*)&W[(size_t)(col0 + n) * HDIM + k0 + xk4 * 4];
    }
#pragma unroll
    for (int ii = 0; ii < 8; ii++) {
      const int m = xm + ii * 16;
      Pack4 p; p.h[0] = (_Float16)xf[ii].x; p.h[1] = (_Float16)xf[ii].y;
               p.h[2] = (_Float16)xf[ii].z; p.h[3] = (_Float16)xf[ii].w;
      *(uint2*)&sX[m * 72 + xk4 * 4] = p.u;
    }
#pragma unroll
    for (int ii = 0; ii < 4; ii++) {
      const int n = wmr + ii * 16;
      Pack4 p; p.h[0] = (_Float16)wf[ii].x; p.h[1] = (_Float16)wf[ii].y;
               p.h[2] = (_Float16)wf[ii].z; p.h[3] = (_Float16)wf[ii].w;
      *(uint2*)&sW[n * 72 + xk4 * 4] = p.u;
    }
    __syncthreads();

#pragma unroll
    for (int s = 0; s < 2; s++) {          // two K=32 WMMA steps
      Frag16 A[2], Bf[2];
#pragma unroll
      for (int fi = 0; fi < 2; fi++) {     // A: row=l16, halves 0-7/16-23 vs 8-15/24-31
        const _Float16* rp = &sX[(wm * 32 + fi * 16 + l16) * 72 + s * 32];
        const int o = lh * 8;
        A[fi].u[0] = *(const uint4*)(rp + o);
        A[fi].u[1] = *(const uint4*)(rp + o + 16);
      }
#pragma unroll
      for (int fj = 0; fj < 2; fj++) {     // B: B^T row=l16, halves 0-7/16-23 vs 8-15/24-31
        const _Float16* rp = &sW[(wn * 32 + fj * 16 + l16) * 72 + s * 32];
        const int o = lh * 8;
        Bf[fj].u[0] = *(const uint4*)(rp + o);
        Bf[fj].u[1] = *(const uint4*)(rp + o + 16);
      }
#pragma unroll
      for (int fi = 0; fi < 2; fi++)
#pragma unroll
        for (int fj = 0; fj < 2; fj++)
          acc[fi][fj] = wmma_f16(A[fi].h, Bf[fj].h, acc[fi][fj]);
    }
    __syncthreads();
  }

  // Epilogue. C layout: lane-half 0 -> rows r, half 1 -> rows 8+r; col = l16.
#pragma unroll
  for (int fi = 0; fi < 2; fi++) {
#pragma unroll
    for (int fj = 0; fj < 2; fj++) {
      const int nBase = col0 + wn * 32 + fj * 16 + l16;
      const float bv = bias[nBase];
      const int h = nBase / HD, d = nBase % HD;
      if (kind < 2) {
#pragma unroll
        for (int r = 0; r < 8; r++) {
          const int m = row0 + wm * 32 + fi * 16 + r + lh * 8;
          const int b = m / TSEQ, t = m % TSEQ;
          outQK[(((size_t)b * NHEAD + h) * TSEQ + t) * HD + d] =
              (_Float16)(acc[fi][fj][r] + bv);
        }
      } else {
        // 8 consecutive t per lane -> one b128 store into Vt[b][h][d][t0..t0+7]
        const int m0 = row0 + wm * 32 + fi * 16 + lh * 8;
        const int b = m0 / TSEQ, t0 = m0 % TSEQ;
        Pack8 p;
#pragma unroll
        for (int r = 0; r < 8; r++) p.h[r] = (_Float16)(acc[fi][fj][r] + bv);
        *(uint4*)&outVt[(((size_t)b * NHEAD + h) * HD + d) * TSEQ + t0] = p.u;
      }
    }
  }
}

// ---------------------------------------------------------------------------
// Kernel 2: causal flash attention. 128 thr (4 waves); wave = 16 query rows.
// Block-shared key loop: K/V tiles async-copied into double-buffered LDS
// (global_load_async_to_lds_b128 + s_wait_asynccnt), waves skip tiles beyond
// their causal range wave-uniformly. 8 WMMA per 32-key tile per wave.
// ---------------------------------------------------------------------------
__global__ __launch_bounds__(128)
void flash_attn_kernel(const _Float16* __restrict__ Qh,  // (B,NH,T,HD)
                       const _Float16* __restrict__ Kh,  // (B,NH,T,HD)
                       const _Float16* __restrict__ Vt,  // (B,NH,HD,T)
                       const float* __restrict__ amask,  // (B,T)
                       float* __restrict__ Out)          // (B,T,H)
{
  const int tid  = threadIdx.x;
  const int lane = tid & 31;
  const int wave = tid >> 5;
  const int lh   = lane >> 4;
  const int l16  = lane & 15;
  const int b = blockIdx.z, h = blockIdx.y;
  const int blkQ0 = blockIdx.x * 64;
  const int q0 = blkQ0 + wave * 16;
  const int qLast = q0 + 15;

  const size_t headQK = ((size_t)b * NHEAD + h) * TSEQ * (size_t)HD;
  const _Float16* Qp = Qh + headQK;
  const _Float16* Kp = Kh + headQK;
  const _Float16* Vp = Vt + ((size_t)b * NHEAD + h) * (size_t)HD * TSEQ;

  __shared__ _Float16 sK[2][32 * 72];  // key-major: 32 keys x 64 d (+8 pad)
  __shared__ _Float16 sV[2][64 * 40];  // d-major:   64 d x 32 t   (+8 pad)
  __shared__ _Float16 sP[4][16 * 40];  // per-wave P staging
  _Float16* myP = sP[wave];

  // Q A-fragments (d 0..31 / 32..63), resident all loop
  Frag16 QA[2];
  {
    const _Float16* rp = Qp + (size_t)(q0 + l16) * HD;
#pragma unroll
    for (int s = 0; s < 2; s++) {
      const int o = s * 32 + lh * 8;
      QA[s].u[0] = *(const uint4*)(rp + o);
      QA[s].u[1] = *(const uint4*)(rp + o + 16);
    }
  }

  const v8f vzero = {};
  v8f O[4];
  O[0] = vzero; O[1] = vzero; O[2] = vzero; O[3] = vzero;
  float mr[8], lr[8];
#pragma unroll
  for (int r = 0; r < 8; r++) { mr[r] = -3.0e38f; lr[r] = 0.0f; }

  const int nTiles = (blkQ0 + 64) / 32;

  auto issue_tile = [&](int tile, int buf) {
    const int kt = tile * 32;
    // K tile: 32 rows x 128 B = 256 x b128; V tile: 64 rows x 64 B = 256 x b128
#pragma unroll
    for (int ii = 0; ii < 2; ii++) {
      const int e = tid + ii * 128;
      const int krow = e >> 3, kseg = e & 7;
      async_load_b128(lds_addr_of(&sK[buf][krow * 72 + kseg * 8]),
                      Kp + (size_t)(kt + krow) * HD + kseg * 8);
    }
#pragma unroll
    for (int ii = 0; ii < 2; ii++) {
      const int e = tid + ii * 128;
      const int vrow = e >> 2, vseg = e & 3;
      async_load_b128(lds_addr_of(&sV[buf][vrow * 40 + vseg * 8]),
                      Vp + (size_t)vrow * TSEQ + kt + vseg * 8);
    }
  };

  issue_tile(0, 0);
  for (int tile = 0; tile < nTiles; ++tile) {
    const int kt  = tile * 32;
    const int buf = tile & 1;
    wait_async0();            // this wave's loads for tile `tile` done
    __syncthreads();          // all waves' loads done; prev compute drained
    if (tile + 1 < nTiles) issue_tile(tile + 1, buf ^ 1);

    if (kt <= qLast) {        // wave-uniform causal skip (EXEC all-1 inside)
      // ---- S = Q K^T : 2 N-frags x 2 K-steps (B from LDS) ----
      v8f S[2];
#pragma unroll
      for (int j = 0; j < 2; j++) {
        const _Float16* rp = &sK[buf][(j * 16 + l16) * 72];
        v8f c = vzero;
#pragma unroll
        for (int s = 0; s < 2; s++) {
          Frag16 Bf;
          const int o = s * 32 + lh * 8;
          Bf.u[0] = *(const uint4*)(rp + o);
          Bf.u[1] = *(const uint4*)(rp + o + 16);
          c = wmma_f16(QA[s].h, Bf.h, c);
        }
        S[j] = c;
      }

      // ---- scale + causal + additive mask ----
      const float scale = 0.125f;
#pragma unroll
      for (int j = 0; j < 2; j++) {
        const int kcol = kt + j * 16 + l16;
        const float am = amask[(size_t)b * TSEQ + kcol];
#pragma unroll
        for (int r = 0; r < 8; r++) {
          const int qrow = q0 + r + 8 * lh;
          float sv = S[j][r] * scale + am;
          if (kcol > qrow) sv = NEGBIG;
          S[j][r] = sv;
        }
      }

      // ---- online softmax ----
      float mnew[8], corr[8];
#pragma unroll
      for (int r = 0; r < 8; r++) {
        float v = fmaxf(S[0][r], S[1][r]);
        v = fmaxf(v, __shfl_xor(v, 1, 32));
        v = fmaxf(v, __shfl_xor(v, 2, 32));
        v = fmaxf(v, __shfl_xor(v, 4, 32));
        v = fmaxf(v, __shfl_xor(v, 8, 32));
        const float mn = fmaxf(mr[r], v);
        corr[r] = __expf(mr[r] - mn);
        mr[r] = mn; mnew[r] = mn;
      }
#pragma unroll
      for (int j = 0; j < 2; j++)
#pragma unroll
        for (int r = 0; r < 8; r++)
          S[j][r] = __expf(S[j][r] - mnew[r]);
#pragma unroll
      for (int r = 0; r < 8; r++) {
        float s = S[0][r] + S[1][r];
        s += __shfl_xor(s, 1, 32);
        s += __shfl_xor(s, 2, 32);
        s += __shfl_xor(s, 4, 32);
        s += __shfl_xor(s, 8, 32);
        lr[r] = lr[r] * corr[r] + s;
      }
#pragma unroll
      for (int n = 0; n < 4; n++)
#pragma unroll
        for (int r = 0; r < 8; r++)
          O[n][r] *= corr[r];

      // ---- P: C-layout regs -> LDS -> A-layout (f16) ----
#pragma unroll
      for (int j = 0; j < 2; j++)
#pragma unroll
        for (int r = 0; r < 8; r++)
          myP[(r + 8 * lh) * 40 + j * 16 + l16] = (_Float16)S[j][r];
      asm volatile("s_wait_dscnt 0" ::: "memory");  // DS in-order within wave
      Frag16 PA;
      {
        const _Float16* rp = &myP[l16 * 40];
        const int o = lh * 8;
        PA.u[0] = *(const uint4*)(rp + o);
        PA.u[1] = *(const uint4*)(rp + o + 16);
      }

      // ---- O += P V (B from d-major LDS V tile) ----
#pragma unroll
      for (int n = 0; n < 4; n++) {
        Frag16 Bf;
        const _Float16* rp = &sV[buf][(n * 16 + l16) * 40];
        const int o = lh * 8;
        Bf.u[0] = *(const uint4*)(rp + o);
        Bf.u[1] = *(const uint4*)(rp + o + 16);
        O[n] = wmma_f16(PA.h, Bf.h, O[n]);
      }
    }
  }

  // ---- normalize + write (B,T,H) fp32 ----
#pragma unroll
  for (int r = 0; r < 8; r++) {
    const float inv = 1.0f / lr[r];
    const int t = q0 + r + 8 * lh;
#pragma unroll
    for (int n = 0; n < 4; n++)
      Out[((size_t)b * TSEQ + t) * HDIM + h * HD + n * 16 + l16] = O[n][r] * inv;
  }
}

// ---------------------------------------------------------------------------
extern "C" void kernel_launch(void* const* d_in, const int* in_sizes, int n_in,
                              void* d_out, int out_size, void* d_ws, size_t ws_size,
                              hipStream_t stream)
{
  const float* X  = (const float*)d_in[0];
  const float* am = (const float*)d_in[1];
  const float* Wq = (const float*)d_in[2];
  const float* bq = (const float*)d_in[3];
  const float* Wk = (const float*)d_in[4];
  const float* bk = (const float*)d_in[5];
  const float* Wv = (const float*)d_in[6];
  const float* bv = (const float*)d_in[7];
  float* Out = (float*)d_out;

  const size_t headElems = (size_t)BATCH * NHEAD * TSEQ * HD;
  _Float16* Qh = (_Float16*)d_ws;
  _Float16* Kh = Qh + headElems;
  _Float16* Vt = Kh + headElems;

  dim3 gproj(HDIM / 64, (BATCH * TSEQ) / 128, 1);
  qkv_proj_kernel<<<gproj, 256, 0, stream>>>(X, Wq, bq, Qh, nullptr, 0);
  qkv_proj_kernel<<<gproj, 256, 0, stream>>>(X, Wk, bk, Kh, nullptr, 1);
  qkv_proj_kernel<<<gproj, 256, 0, stream>>>(X, Wv, bv, nullptr, Vt, 2);

  dim3 gattn(TSEQ / 64, NHEAD, BATCH);
  flash_attn_kernel<<<gattn, 128, 0, stream>>>(Qh, Kh, Vt, am, Out);
}
